// CriticGNN_40630390620300
// MI455X (gfx1250) — matrix operand order repaired and checked by reading, not compiled, not run on hardware
//
#include <hip/hip_runtime.h>

// -------- problem constants (from reference) --------
#define NN   100000
#define EE   3200000
#define BB   64
#define FIN  64
#define EIN  16
#define ADIM 13
#define HH   32
#define OUTC 64
#define NSTRIPS (NN/16)        // 6250, exact
#define EPS_GEN 1e-7f
#define EPS_BN  1e-5f

typedef __attribute__((ext_vector_type(16))) _Float16 v16h;
typedef __attribute__((ext_vector_type(8)))  float    v8f;

// ---------------- utility kernels ----------------
__global__ void gnn_zero_kernel(float* __restrict__ p, int n) {
  int i = blockIdx.x * blockDim.x + threadIdx.x;
  int stride = gridDim.x * blockDim.x;
  for (; i < n; i += stride) p[i] = 0.0f;
}

// Pack W[KC,NC] (f32 row-major) into WMMA B fragments, fragment-lane order:
// frag = kb*(NC/16)+t ; within fragment: [lane][e] contiguous (32B per lane).
// B layout: lane -> col n = t*16+(lane&15), K = kb*32 + (lane>>4)*8 + (e&7) + (e>>3)*16.
__global__ void gnn_pack_b_kernel(const float* __restrict__ w, _Float16* __restrict__ dst,
                                  int KC, int NC) {
  int i = blockIdx.x * blockDim.x + threadIdx.x;
  if (i >= KC * NC) return;
  int e    = i & 15;
  int lane = (i >> 4) & 31;
  int frag = i >> 9;
  int tcols = NC >> 4;
  int kb = frag / tcols, t = frag % tcols;
  int k = kb * 32 + ((lane >> 4) << 3) + (e & 7) + ((e >> 3) << 4);
  int n = t * 16 + (lane & 15);
  dst[i] = (_Float16)w[k * NC + n];
}

// ---------------- fused GEMM: out[N,NC] = act(in[N,KC]) @ W + b (+BN stats) ----------------
// One wave = one 16-row strip. A staged to LDS in fragment order (BN+ReLU+f16 on the fly),
// read back as aligned 32B vectors; B fragments are 32B contiguous per lane (pre-packed).
template <int KC, int NC>
__global__ __launch_bounds__(256) void gnn_gemm_kernel(
    const float* __restrict__ in,
    const float* __restrict__ bnscale, const float* __restrict__ bnshift,
    const _Float16* __restrict__ wpk, const float* __restrict__ bias,
    float* __restrict__ out, float* __restrict__ stats) {
  __shared__ alignas(32) _Float16 sA[8][KC * 16];
  int wave = threadIdx.x >> 5, lane = threadIdx.x & 31;
  int strip = blockIdx.x * 8 + wave;
  bool active = strip < NSTRIPS;
  int row0 = strip * 16;
  const bool useBN = (bnscale != nullptr);
  if (active) {
    // float4 coalesced loads; scatter into A-fragment layout:
    // row r -> lane (r + 16*grp), col c -> kb=c/32, grp=bit3(c&31), e per ISA A map.
    for (int base = lane * 4; base < 16 * KC; base += 128) {
      int r = base / KC, c = base % KC;           // c % 4 == 0, same row for 4 elems
      const float4 v4 = *reinterpret_cast<const float4*>(&in[(row0 + r) * KC + c]);
      float vv[4] = {v4.x, v4.y, v4.z, v4.w};
#pragma unroll
      for (int j = 0; j < 4; ++j) {
        int cc = c + j;
        float v = vv[j];
        if (useBN) v = fmaxf(v * bnscale[cc] + bnshift[cc], 0.0f);
        int kb  = cc >> 5;
        int kl  = cc & 31;
        int grp = (kl >> 3) & 1;
        int kk  = kl - (grp << 3);                // in {0..7} u {16..23}
        int e   = (kk < 8) ? kk : (kk - 8);
        sA[wave][(kb << 9) + ((r + (grp << 4)) << 4) + e] = (_Float16)v;
      }
    }
  }
  __syncthreads();
  if (!active) return;
  int hi = lane >> 4, nl = lane & 15;
  v16h a0 = *reinterpret_cast<const v16h*>(&sA[wave][lane * 16]);
  v16h a1 = {};
  if (KC == 64) a1 = *reinterpret_cast<const v16h*>(&sA[wave][512 + lane * 16]);
#pragma unroll
  for (int t = 0; t < NC / 16; ++t) {
    v8f c = {};
    v16h b0 = *reinterpret_cast<const v16h*>(&wpk[(t << 9) + lane * 16]);
    c = __builtin_amdgcn_wmma_f32_16x16x32_f16(false, a0, false, b0, (short)0, c, false, false);
    if (KC == 64) {
      v16h b1 = *reinterpret_cast<const v16h*>(&wpk[((NC / 16 + t) << 9) + lane * 16]);
      c = __builtin_amdgcn_wmma_f32_16x16x32_f16(false, a1, false, b1, (short)0, c, false, false);
    }
    int col = t * 16 + nl;
    float bc = bias[col];
    float ssum = 0.0f, ssq = 0.0f;
#pragma unroll
    for (int r = 0; r < 8; ++r) {
      float v = c[r] + bc;
      out[(row0 + r + hi * 8) * NC + col] = v;
      ssum += v;
      ssq  += v * v;
    }
    if (stats) {
      ssum += __shfl_xor(ssum, 16);
      ssq  += __shfl_xor(ssq, 16);
      if (hi == 0) {
        atomicAdd(&stats[col], ssum);
        atomicAdd(&stats[NC + col], ssq);
      }
    }
  }
}

// ---------------- edge passes ----------------
// phase 0: msg -> atomicMax bits into mx      (msg > 0, so uint order == float order)
// phase 1: e=exp(msg-mx); atomicAdd e -> sm, msg*e -> wsum
__global__ __launch_bounds__(256) void gnn_edge_kernel(
    const int* __restrict__ ei, const float* __restrict__ eattr,
    const float* __restrict__ ew, const float* __restrict__ eb,
    const float* __restrict__ h, unsigned int* __restrict__ mx,
    float* __restrict__ sm, float* __restrict__ wsum, int phase) {
  __shared__ float sW[EIN * HH];   // 512 floats
  __shared__ float sE[8][EIN];
  int t = threadIdx.x;
  int e0 = blockIdx.x * 8;
  sW[t]       = ew[t];
  sW[t + 256] = ew[t + 256];
  if (t < 128) {
    int e = e0 + (t >> 4);
    sE[t >> 4][t & 15] = (e < EE) ? eattr[e * EIN + (t & 15)] : 0.0f;
  }
  if (t == 0 && e0 + 16 <= EE)
    __builtin_prefetch(&eattr[(e0 + 8) * EIN], 0, 3);   // global_prefetch next tile
  __syncthreads();
  int el = t >> 5, c = t & 31;
  int e = e0 + el;
  if (e >= EE) return;
  float acc = eb[c];
#pragma unroll
  for (int k = 0; k < EIN; ++k) acc += sE[el][k] * sW[k * HH + c];
  int s = ei[e], d = ei[EE + e];
  float m = fmaxf(h[s * HH + c] + acc, 0.0f) + EPS_GEN;
  int idx = d * HH + c;
  if (phase == 0) {
    atomicMax(&mx[idx], __float_as_uint(m));
  } else {
    float mv = __uint_as_float(mx[idx]);
    float ex = __expf(m - mv);
    atomicAdd(&sm[idx], ex);
    atomicAdd(&wsum[idx], m * ex);
  }
}

// ---------------- aggregation: act = wsum/sm + h ----------------
__global__ void gnn_agg_kernel(const float* __restrict__ sm, const float* __restrict__ wsum,
                               const float* __restrict__ h, float* __restrict__ act) {
  int i = blockIdx.x * blockDim.x + threadIdx.x;
  if (i < NN * HH) {
    float s = sm[i];
    act[i] = ((s > 0.0f) ? (wsum[i] / s) : 0.0f) + h[i];
  }
}

// ---------------- fold BN batch stats into per-channel scale/shift ----------------
__global__ void gnn_bnfin_kernel(const float* __restrict__ stats, const float* __restrict__ g,
                                 const float* __restrict__ b, float* __restrict__ sc,
                                 float* __restrict__ sh) {
  int c = threadIdx.x;
  if (c < OUTC) {
    float mean = stats[c] * (1.0f / (float)NN);
    float var  = stats[OUTC + c] * (1.0f / (float)NN) - mean * mean;
    float s = g[c] * rsqrtf(var + EPS_BN);
    sc[c] = s;
    sh[c] = b[c] - mean * s;
  }
}

// ---------------- global mean pool (sums + counts via atomics) ----------------
__global__ void gnn_pool_kernel(const float* __restrict__ nodeout, const int* __restrict__ batch,
                                float* __restrict__ pool, float* __restrict__ cnt) {
  int i = blockIdx.x * blockDim.x + threadIdx.x;
  if (i < NN * OUTC) {
    int n = i >> 6, c = i & 63;
    int g = batch[n];
    atomicAdd(&pool[g * OUTC + c], nodeout[i]);
    if (c == 0) atomicAdd(&cnt[g], 1.0f);
  }
}

// ---------------- policy head: one thread per graph ----------------
__global__ void gnn_head_kernel(const float* __restrict__ pool, const float* __restrict__ cnt,
                                const float* __restrict__ action,
                                const float* __restrict__ pin_w, const float* __restrict__ pin_b,
                                const float* __restrict__ ph_w, const float* __restrict__ ph_b,
                                const float* __restrict__ po_w, const float* __restrict__ po_b,
                                float* __restrict__ out) {
  int b = threadIdx.x;
  if (b >= BB) return;
  float inv = 1.0f / fmaxf(cnt[b], 1.0f);
  float fp[16];
#pragma unroll
  for (int j = 0; j < 16; ++j) {
    float acc = pin_b[j];
    for (int c = 0; c < OUTC; ++c) acc += (pool[b * OUTC + c] * inv) * pin_w[c * 16 + j];
    fp[j] = fmaxf(acc, 0.0f);
  }
  float hd[10];
#pragma unroll
  for (int j = 0; j < 10; ++j) {
    float acc = ph_b[j];
#pragma unroll
    for (int i = 0; i < 16; ++i) acc += fp[i] * ph_w[i * 10 + j];
#pragma unroll
    for (int i = 0; i < ADIM; ++i) acc += action[b * ADIM + i] * ph_w[(16 + i) * 10 + j];
    hd[j] = fmaxf(acc, 0.0f);
  }
  float o = po_b[0];
#pragma unroll
  for (int j = 0; j < 10; ++j) o += hd[j] * po_w[j];
  out[b] = o;
}

// ---------------- launcher ----------------
extern "C" void kernel_launch(void* const* d_in, const int* in_sizes, int n_in,
                              void* d_out, int out_size, void* d_ws, size_t ws_size,
                              hipStream_t stream) {
  const float* x      = (const float*)d_in[0];
  const int*   ei     = (const int*)d_in[1];
  const float* eattr  = (const float*)d_in[2];
  const int*   batch  = (const int*)d_in[3];
  const float* action = (const float*)d_in[4];
  const float* node_w = (const float*)d_in[5];
  const float* node_b = (const float*)d_in[6];
  const float* edge_w = (const float*)d_in[7];
  const float* edge_b = (const float*)d_in[8];
  const float* w1 = (const float*)d_in[9];   const float* b1 = (const float*)d_in[10];
  const float* g1 = (const float*)d_in[11];  const float* be1 = (const float*)d_in[12];
  const float* w2 = (const float*)d_in[13];  const float* b2 = (const float*)d_in[14];
  const float* g2 = (const float*)d_in[15];  const float* be2 = (const float*)d_in[16];
  const float* w3 = (const float*)d_in[17];  const float* b3 = (const float*)d_in[18];
  const float* g3 = (const float*)d_in[19];  const float* be3 = (const float*)d_in[20];
  const float* w4 = (const float*)d_in[21];  const float* b4 = (const float*)d_in[22];
  const float* pin_w = (const float*)d_in[23]; const float* pin_b = (const float*)d_in[24];
  const float* ph_w  = (const float*)d_in[25]; const float* ph_b  = (const float*)d_in[26];
  const float* po_w  = (const float*)d_in[27]; const float* po_b  = (const float*)d_in[28];

  // workspace arena (floats); requires ~64.1 MB
  float* ws = (float*)d_ws;
  float* R0 = ws;                                   // N*64
  float* R1 = ws + (size_t)NN * 64;                 // N*64
  float* R2 = R1 + (size_t)NN * 64;                 // N*32  (h)
  float* S  = R2 + (size_t)NN * 32;                 // smalls
  float* stats1 = S;              float* stats2 = stats1 + 128; float* stats3 = stats2 + 128;
  float* bnsc1 = stats3 + 128;    float* bnsh1 = bnsc1 + 64;
  float* bnsc2 = bnsh1 + 64;      float* bnsh2 = bnsc2 + 64;
  float* bnsc3 = bnsh2 + 64;      float* bnsh3 = bnsc3 + 64;
  float* pool  = bnsh3 + 64;                        // 64*64
  float* cnt   = pool + BB * OUTC;                  // 64
  // packed f16 weight fragments, 32B aligned
  _Float16* wn16 = (_Float16*)((((uintptr_t)(cnt + BB)) + 31) & ~(uintptr_t)31);
  _Float16* w116 = wn16 + 64 * 32;                  // 4 frags  (KC=64, NC=32)
  _Float16* w216 = w116 + 32 * 64;                  // 4 frags  (KC=32, NC=64)
  _Float16* w316 = w216 + 64 * 64;                  // 8 frags
  _Float16* w416 = w316 + 64 * 64;                  // 8 frags

  float* h    = R2;
  float* mx   = R0;                                 // uint bits, init 0 (msg > 0)
  float* sm   = R0 + (size_t)NN * 32;
  float* wsum = R1;
  float* act0 = R1 + (size_t)NN * 32;
  float* z1 = R0; float* z2 = R1; float* z3 = R0; float* nodeout = R1;

  // zero: mx+sm (all of R0) + wsum (R1 lower half); plus stats/pool/cnt
  gnn_zero_kernel<<<2048, 256, 0, stream>>>(R0, NN * 96);
  gnn_zero_kernel<<<20, 256, 0, stream>>>(S, 384 + 6 * 64 + BB * OUTC + BB);

  // pack weights into WMMA B-fragment order (f16)
  gnn_pack_b_kernel<<<(64 * 32 + 255) / 256, 256, 0, stream>>>(node_w, wn16, 64, 32);
  gnn_pack_b_kernel<<<(32 * 64 + 255) / 256, 256, 0, stream>>>(w1, w116, 32, 64);
  gnn_pack_b_kernel<<<(64 * 64 + 255) / 256, 256, 0, stream>>>(w2, w216, 64, 64);
  gnn_pack_b_kernel<<<(64 * 64 + 255) / 256, 256, 0, stream>>>(w3, w316, 64, 64);
  gnn_pack_b_kernel<<<(64 * 64 + 255) / 256, 256, 0, stream>>>(w4, w416, 64, 64);

  int gemmBlocks = (NSTRIPS + 7) / 8;   // 782
  // node encoder: h = x @ node_w + node_b
  gnn_gemm_kernel<64, 32><<<gemmBlocks, 256, 0, stream>>>(x, nullptr, nullptr, wn16, node_b,
                                                          h, nullptr);

  int edgeBlocks = (EE + 7) / 8;        // 400000
  gnn_edge_kernel<<<edgeBlocks, 256, 0, stream>>>(ei, eattr, edge_w, edge_b, h,
                                                  (unsigned int*)mx, nullptr, nullptr, 0);
  gnn_edge_kernel<<<edgeBlocks, 256, 0, stream>>>(ei, eattr, edge_w, edge_b, h,
                                                  (unsigned int*)mx, sm, wsum, 1);

  gnn_agg_kernel<<<(NN * HH + 255) / 256, 256, 0, stream>>>(sm, wsum, h, act0);

  gnn_gemm_kernel<32, 64><<<gemmBlocks, 256, 0, stream>>>(act0, nullptr, nullptr, w116, b1,
                                                          z1, stats1);
  gnn_bnfin_kernel<<<1, 64, 0, stream>>>(stats1, g1, be1, bnsc1, bnsh1);
  gnn_gemm_kernel<64, 64><<<gemmBlocks, 256, 0, stream>>>(z1, bnsc1, bnsh1, w216, b2,
                                                          z2, stats2);
  gnn_bnfin_kernel<<<1, 64, 0, stream>>>(stats2, g2, be2, bnsc2, bnsh2);
  gnn_gemm_kernel<64, 64><<<gemmBlocks, 256, 0, stream>>>(z2, bnsc2, bnsh2, w316, b3,
                                                          z3, stats3);
  gnn_bnfin_kernel<<<1, 64, 0, stream>>>(stats3, g3, be3, bnsc3, bnsh3);
  gnn_gemm_kernel<64, 64><<<gemmBlocks, 256, 0, stream>>>(z3, bnsc3, bnsh3, w416, b4,
                                                          nodeout, nullptr);

  gnn_pool_kernel<<<(NN * OUTC + 255) / 256, 256, 0, stream>>>(nodeout, batch, pool, cnt);
  gnn_head_kernel<<<1, 64, 0, stream>>>(pool, cnt, action, pin_w, pin_b, ph_w, ph_b,
                                        po_w, po_b, (float*)d_out);
}
